// SparseMixtureOfExperts_2422361555261
// MI455X (gfx1250) — matrix operand order repaired and checked by reading, not compiled, not run on hardware
//
#include <hip/hip_runtime.h>
#include <hip/hip_bf16.h>
#include <math.h>

#define DIMK 1024
#define HIDK 2048
#define OUTK 1024
#define NEXP 8
#define NTOK 4096
#define PADROWS (2 * NTOK + NEXP * 16)   // 8320, multiple of 16

typedef __attribute__((ext_vector_type(16))) __bf16 v16bf;
typedef __attribute__((ext_vector_type(8)))  float  v8f;

union FragAB { v16bf v; uint4 q[2]; };
union FragC  { v8f  v; float f[8]; };

// A-matrix 16x32 bf16 fragment (row-major, contiguous K at rowp):
// lanes 0-15: K{0..7, 16..23}; lanes 16-31: K{8..15, 24..31}
__device__ __forceinline__ v16bf load_frag_a16(const __bf16* rowp, int lane) {
  const uint4* p = (const uint4*)rowp;
  FragAB u;
  int hi = (lane >> 4) & 1;
  u.q[0] = p[hi];        // K 0..7   | 8..15
  u.q[1] = p[2 + hi];    // K 16..23 | 24..31
  return u.v;
}
// B-matrix 32x16 bf16 fragment; colp = contiguous K-run of column n (B^T row):
// lanes 0-15: K{0..7, 16..23}; lanes 16-31: K{8..15, 24..31}
__device__ __forceinline__ v16bf load_frag_b16(const __bf16* colp, int lane) {
  const uint4* p = (const uint4*)colp;
  FragAB u;
  int hi = (lane >> 4) & 1;
  u.q[0] = p[hi];
  u.q[1] = p[2 + hi];
  return u.v;
}
__device__ __forceinline__ v8f wmma_bf16(v16bf a, v16bf b, v8f c) {
  return __builtin_amdgcn_wmma_f32_16x16x32_bf16(false, a, false, b, (short)0, c,
                                                 false, false);
}

// ---------------- conversion / transpose prepass ----------------
__global__ __launch_bounds__(256) void k_cvt_bf16(const float* __restrict__ src,
                                                  __bf16* __restrict__ dst, size_t n) {
  for (size_t i = (size_t)blockIdx.x * blockDim.x + threadIdx.x; i < n;
       i += (size_t)gridDim.x * blockDim.x)
    dst[i] = (__bf16)src[i];
}

__global__ __launch_bounds__(256) void k_zero_bf16(__bf16* __restrict__ dst, int n) {
  int i = blockIdx.x * blockDim.x + threadIdx.x;
  if (i < n) dst[i] = (__bf16)0.0f;
}

// src [E][K][N] fp32 -> dst [E][N][K] bf16  (B^T layout for WMMA B fragments)
__global__ __launch_bounds__(256) void k_cvt_transpose_bf16(const float* __restrict__ src,
                                                            __bf16* __restrict__ dst,
                                                            int K, int N, int E) {
  size_t kn = (size_t)K * N;
  size_t total = kn * E;
  for (size_t i = (size_t)blockIdx.x * blockDim.x + threadIdx.x; i < total;
       i += (size_t)gridDim.x * blockDim.x) {
    size_t e = i / kn;
    size_t r = i - e * kn;
    size_t k = r / (size_t)N;
    size_t n = r - k * (size_t)N;
    dst[e * kn + n * (size_t)K + k] = (__bf16)src[i];
  }
}

// ---------------- gating: one wave per token ----------------
__global__ __launch_bounds__(128) void k_gate(const float* __restrict__ X,
                                              const float* __restrict__ Wg,
                                              const float* __restrict__ bg,
                                              int* __restrict__ topk_idx,
                                              float* __restrict__ topk_w) {
  int wave = threadIdx.x >> 5;
  int lane = threadIdx.x & 31;
  int t = blockIdx.x * 4 + wave;
  if (t >= NTOK) return;
  float acc[NEXP];
#pragma unroll
  for (int e = 0; e < NEXP; e++) acc[e] = 0.f;
  const float* xr = X + (size_t)t * DIMK;
  for (int d = lane; d < DIMK; d += 32) {
    float xv = xr[d];
    const float* wr = Wg + (size_t)d * NEXP;
#pragma unroll
    for (int e = 0; e < NEXP; e++) acc[e] += xv * wr[e];
  }
#pragma unroll
  for (int e = 0; e < NEXP; e++) {
    float v = acc[e];
#pragma unroll
    for (int off = 16; off > 0; off >>= 1) v += __shfl_xor(v, off, 32);
    acc[e] = v + bg[e];
  }
  if (lane == 0) {
    float m = acc[0];
#pragma unroll
    for (int e = 1; e < NEXP; e++) m = fmaxf(m, acc[e]);
    float p[NEXP];
    float s = 0.f;
#pragma unroll
    for (int e = 0; e < NEXP; e++) { p[e] = __expf(acc[e] - m); s += p[e]; }
    float inv = 1.f / s;
#pragma unroll
    for (int e = 0; e < NEXP; e++) p[e] *= inv;
    // top-1/top-2 with largest-index tie break (matches stable ascending argsort)
    int i1 = 0; float v1 = p[0];
#pragma unroll
    for (int e = 1; e < NEXP; e++) if (p[e] >= v1) { v1 = p[e]; i1 = e; }
    int i2 = (i1 == 0) ? 1 : 0; float v2 = p[i2];
#pragma unroll
    for (int e = 0; e < NEXP; e++)
      if (e != i1 && p[e] >= v2) { v2 = p[e]; i2 = e; }
    topk_idx[t * 2 + 0] = i1; topk_w[t * 2 + 0] = v1;
    topk_idx[t * 2 + 1] = i2; topk_w[t * 2 + 1] = v2;
  }
}

// ---------------- deterministic routing (single block, no atomics) ----------------
__global__ __launch_bounds__(64) void k_route(const int* __restrict__ topk_idx,
                                              int* __restrict__ row_token,
                                              int* __restrict__ row_expert,
                                              int* __restrict__ row_of) {
  __shared__ int cnt[NEXP];
  __shared__ int base[NEXP];
  __shared__ int padded_total;
  int tid = threadIdx.x;
  if (tid < NEXP) {
    int c = 0;
    for (int i = 0; i < 2 * NTOK; i++)
      if (topk_idx[i] == tid) c++;
    cnt[tid] = c;
  }
  __syncthreads();
  if (tid == 0) {
    int b = 0;
    for (int e = 0; e < NEXP; e++) { base[e] = b; b += (cnt[e] + 15) & ~15; }
    padded_total = b;
  }
  __syncthreads();
  if (tid < NEXP) {
    int pos = base[tid];
    for (int i = 0; i < 2 * NTOK; i++) {
      if (topk_idx[i] == tid) {
        row_token[pos] = i >> 1;   // token id
        row_expert[pos] = tid;
        row_of[i] = pos;
        pos++;
      }
    }
    int end = base[tid] + ((cnt[tid] + 15) & ~15);
    for (; pos < end; pos++) { row_token[pos] = -1; row_expert[pos] = tid; }
  }
  __syncthreads();
  if (tid == 0) {
    for (int pos = padded_total; pos < PADROWS; pos++) {
      row_token[pos] = -1;
      row_expert[pos] = 0;
    }
  }
}

// ---- GEMM1: gathered X @ {W1,W3}, 16x32 tile/wave, fused bias + silu(h1*h3) ----
__global__ __launch_bounds__(256) void k_ffn1(const __bf16* __restrict__ Xbf,
                                              const __bf16* __restrict__ Zrow,
                                              const __bf16* __restrict__ W1T,
                                              const __bf16* __restrict__ W3T,
                                              const float* __restrict__ b1,
                                              const float* __restrict__ b3,
                                              const int* __restrict__ row_token,
                                              const int* __restrict__ row_expert,
                                              __bf16* __restrict__ H) {
  int wave = threadIdx.x >> 5;
  int lane = threadIdx.x & 31;
  int rowbase = blockIdx.y * 16;
  int e = row_expert[rowbase];
  int n0 = blockIdx.x * 256 + wave * 32 + (lane & 15);
  int tok = row_token[rowbase + (lane & 15)];
  // padded rows read a pre-zeroed row: no per-iteration masking in the hot loop
  const __bf16* arow = (tok >= 0) ? (Xbf + (size_t)tok * DIMK) : Zrow;
  const __bf16* w1a = W1T + ((size_t)e * HIDK + n0) * DIMK;
  const __bf16* w1b = w1a + (size_t)16 * DIMK;
  const __bf16* w3a = W3T + ((size_t)e * HIDK + n0) * DIMK;
  const __bf16* w3b = w3a + (size_t)16 * DIMK;
  v8f acc1a = {0.f, 0.f, 0.f, 0.f, 0.f, 0.f, 0.f, 0.f};
  v8f acc1b = acc1a, acc3a = acc1a, acc3b = acc1a;
  for (int k0 = 0; k0 < DIMK; k0 += 32) {
    __builtin_prefetch(w1a + k0 + 256, 0, 3);
    __builtin_prefetch(w3a + k0 + 256, 0, 3);
    v16bf a = load_frag_a16(arow + k0, lane);
    v16bf b1a = load_frag_b16(w1a + k0, lane);
    v16bf b1b = load_frag_b16(w1b + k0, lane);
    v16bf b3a = load_frag_b16(w3a + k0, lane);
    v16bf b3b = load_frag_b16(w3b + k0, lane);
    acc1a = wmma_bf16(a, b1a, acc1a);
    acc1b = wmma_bf16(a, b1b, acc1b);
    acc3a = wmma_bf16(a, b3a, acc3a);
    acc3b = wmma_bf16(a, b3b, acc3b);
  }
  float bb1a = b1[e * HIDK + n0], bb1b = b1[e * HIDK + n0 + 16];
  float bb3a = b3[e * HIDK + n0], bb3b = b3[e * HIDK + n0 + 16];
  FragC u1a, u1b, u3a, u3b;
  u1a.v = acc1a; u1b.v = acc1b; u3a.v = acc3a; u3b.v = acc3b;
  int rb = rowbase + ((lane & 16) ? 8 : 0);
#pragma unroll
  for (int i = 0; i < 8; i++) {
    float pa = (u1a.f[i] + bb1a) * (u3a.f[i] + bb3a);   // h1 * h3
    float pb = (u1b.f[i] + bb1b) * (u3b.f[i] + bb3b);
    float ha = pa / (1.0f + __expf(-pa));               // silu(h1*h3)
    float hb = pb / (1.0f + __expf(-pb));
    H[(size_t)(rb + i) * HIDK + n0] = (__bf16)ha;
    H[(size_t)(rb + i) * HIDK + n0 + 16] = (__bf16)hb;
  }
}

// ---- GEMM2: H @ W2 + b2 -> fp32 expert outputs, 16x32 tile/wave ----
__global__ __launch_bounds__(256) void k_ffn2(const __bf16* __restrict__ H,
                                              const __bf16* __restrict__ W2T,
                                              const float* __restrict__ b2,
                                              const int* __restrict__ row_expert,
                                              float* __restrict__ Y) {
  int wave = threadIdx.x >> 5;
  int lane = threadIdx.x & 31;
  int rowbase = blockIdx.y * 16;
  int e = row_expert[rowbase];
  int n0 = blockIdx.x * 256 + wave * 32 + (lane & 15);
  const __bf16* arow = H + (size_t)(rowbase + (lane & 15)) * HIDK;
  const __bf16* w2a = W2T + ((size_t)e * OUTK + n0) * HIDK;
  const __bf16* w2b = w2a + (size_t)16 * HIDK;
  v8f acca = {0.f, 0.f, 0.f, 0.f, 0.f, 0.f, 0.f, 0.f};
  v8f accb = acca;
  for (int k0 = 0; k0 < HIDK; k0 += 32) {
    __builtin_prefetch(w2a + k0 + 256, 0, 3);
    v16bf a = load_frag_a16(arow + k0, lane);
    v16bf ba = load_frag_b16(w2a + k0, lane);
    v16bf bb = load_frag_b16(w2b + k0, lane);
    acca = wmma_bf16(a, ba, acca);
    accb = wmma_bf16(a, bb, accb);
  }
  float bba = b2[e * OUTK + n0], bbb = b2[e * OUTK + n0 + 16];
  FragC ua, ub;
  ua.v = acca; ub.v = accb;
  int rb = rowbase + ((lane & 16) ? 8 : 0);
#pragma unroll
  for (int i = 0; i < 8; i++) {
    Y[(size_t)(rb + i) * OUTK + n0] = ua.f[i] + bba;
    Y[(size_t)(rb + i) * OUTK + n0 + 16] = ub.f[i] + bbb;
  }
}

// ---------------- mix top-2 expert outputs with gate weights ----------------
__global__ __launch_bounds__(256) void k_mix(const float* __restrict__ Y,
                                             const float* __restrict__ topk_w,
                                             const int* __restrict__ row_of,
                                             __bf16* __restrict__ M) {
  int t = blockIdx.x;
  int o = blockIdx.y * 256 + threadIdx.x;
  int r0 = row_of[t * 2 + 0];
  int r1 = row_of[t * 2 + 1];
  float w0 = topk_w[t * 2 + 0];
  float w1 = topk_w[t * 2 + 1];
  float v = w0 * Y[(size_t)r0 * OUTK + o] + w1 * Y[(size_t)r1 * OUTK + o];
  M[(size_t)t * OUTK + o] = (__bf16)v;
}

// ---- final GEMM: mixed @ Wf + bf -> fp32 d_out, 16x32 tile/wave ----
__global__ __launch_bounds__(256) void k_final(const __bf16* __restrict__ M,
                                               const __bf16* __restrict__ WfT,
                                               const float* __restrict__ bf_,
                                               float* __restrict__ Out) {
  int wave = threadIdx.x >> 5;
  int lane = threadIdx.x & 31;
  int rowbase = blockIdx.y * 16;
  int n0 = blockIdx.x * 256 + wave * 32 + (lane & 15);
  const __bf16* arow = M + (size_t)(rowbase + (lane & 15)) * OUTK;
  const __bf16* wa = WfT + (size_t)n0 * OUTK;
  const __bf16* wb = wa + (size_t)16 * OUTK;
  v8f acca = {0.f, 0.f, 0.f, 0.f, 0.f, 0.f, 0.f, 0.f};
  v8f accb = acca;
  for (int k0 = 0; k0 < OUTK; k0 += 32) {
    __builtin_prefetch(wa + k0 + 256, 0, 3);
    v16bf a = load_frag_a16(arow + k0, lane);
    v16bf ba = load_frag_b16(wa + k0, lane);
    v16bf bb = load_frag_b16(wb + k0, lane);
    acca = wmma_bf16(a, ba, acca);
    accb = wmma_bf16(a, bb, accb);
  }
  float bba = bf_[n0], bbb = bf_[n0 + 16];
  FragC ua, ub;
  ua.v = acca; ub.v = accb;
  int rb = rowbase + ((lane & 16) ? 8 : 0);
#pragma unroll
  for (int i = 0; i < 8; i++) {
    Out[(size_t)(rb + i) * OUTK + n0] = ua.f[i] + bba;
    Out[(size_t)(rb + i) * OUTK + n0 + 16] = ub.f[i] + bbb;
  }
}

extern "C" void kernel_launch(void* const* d_in, const int* in_sizes, int n_in,
                              void* d_out, int out_size, void* d_ws, size_t ws_size,
                              hipStream_t stream) {
  const float* X   = (const float*)d_in[0];
  const float* W1  = (const float*)d_in[1];
  const float* b1  = (const float*)d_in[2];
  const float* W3  = (const float*)d_in[3];
  const float* b3  = (const float*)d_in[4];
  const float* W2  = (const float*)d_in[5];
  const float* b2  = (const float*)d_in[6];
  const float* Wg  = (const float*)d_in[7];
  const float* bg  = (const float*)d_in[8];
  const float* Wf  = (const float*)d_in[9];
  const float* bf_ = (const float*)d_in[10];
  float* Out = (float*)d_out;

  // workspace carve (256B aligned)
  char* p = (char*)d_ws;
  auto take = [&](size_t bytes) -> char* {
    char* r = p;
    p += (bytes + 255) & ~(size_t)255;
    return r;
  };
  __bf16* Xbf  = (__bf16*)take((size_t)NTOK * DIMK * 2);
  __bf16* W1T  = (__bf16*)take((size_t)NEXP * DIMK * HIDK * 2);
  __bf16* W3T  = (__bf16*)take((size_t)NEXP * DIMK * HIDK * 2);
  __bf16* W2T  = (__bf16*)take((size_t)NEXP * HIDK * OUTK * 2);
  __bf16* WfT  = (__bf16*)take((size_t)OUTK * OUTK * 2);
  __bf16* Hbuf = (__bf16*)take((size_t)PADROWS * HIDK * 2);
  float*  Yexp = (float*)take((size_t)PADROWS * OUTK * 4);
  __bf16* Mix  = (__bf16*)take((size_t)NTOK * OUTK * 2);
  __bf16* Zrow = (__bf16*)take((size_t)DIMK * 2);
  int*   topk_idx = (int*)take((size_t)NTOK * 2 * 4);
  float* topk_w   = (float*)take((size_t)NTOK * 2 * 4);
  int*   row_token  = (int*)take((size_t)PADROWS * 4);
  int*   row_expert = (int*)take((size_t)PADROWS * 4);
  int*   row_of     = (int*)take((size_t)NTOK * 2 * 4);

  // 1) precision conversion / weight transposes (one-shot, HBM-bound)
  k_cvt_bf16<<<2048, 256, 0, stream>>>(X, Xbf, (size_t)NTOK * DIMK);
  k_zero_bf16<<<(DIMK + 255) / 256, 256, 0, stream>>>(Zrow, DIMK);
  k_cvt_transpose_bf16<<<4096, 256, 0, stream>>>(W1, W1T, DIMK, HIDK, NEXP);
  k_cvt_transpose_bf16<<<4096, 256, 0, stream>>>(W3, W3T, DIMK, HIDK, NEXP);
  k_cvt_transpose_bf16<<<4096, 256, 0, stream>>>(W2, W2T, HIDK, OUTK, NEXP);
  k_cvt_transpose_bf16<<<1024, 256, 0, stream>>>(Wf, WfT, OUTK, OUTK, 1);

  // 2) gating + 3) deterministic routing
  k_gate<<<NTOK / 4, 128, 0, stream>>>(X, Wg, bg, topk_idx, topk_w);
  k_route<<<1, 64, 0, stream>>>(topk_idx, row_token, row_expert, row_of);

  // 4) expert GEMM phase 1 (dual W1/W3, 16x32 tiles, fused silu)
  k_ffn1<<<dim3(HIDK / 256, PADROWS / 16), 256, 0, stream>>>(
      Xbf, Zrow, W1T, W3T, b1, b3, row_token, row_expert, Hbuf);

  // 5) expert GEMM phase 2
  k_ffn2<<<dim3(OUTK / 256, PADROWS / 16), 256, 0, stream>>>(
      Hbuf, W2T, b2, row_expert, Yexp);

  // 6) top-2 mixing
  k_mix<<<dim3(NTOK, OUTK / 256), 256, 0, stream>>>(Yexp, topk_w, row_of, Mix);

  // 7) final projection
  k_final<<<dim3(OUTK / 256, NTOK / 16), 256, 0, stream>>>(Mix, WfT, bf_, Out);

  (void)in_sizes; (void)n_in; (void)out_size; (void)ws_size;
}
